// CrossAttention_29807073034826
// MI455X (gfx1250) — compile-verified, matrix operand order checked by reading, not run on hardware
//
#include <hip/hip_runtime.h>
#include <hip/hip_bf16.h>

// ---------------------------------------------------------------------------
// Cross-attention for MI455X (gfx1250, wave32, WMMA bf16 16x16x32 + TDM).
// Compute-bound (~940 FLOP/byte) -> everything routed through v_wmma.
// Trick #1: Y = X @ W^T means W fragments load with the SAME pattern as A
//   fragments from row-major X (ISA 7.12.2) -> no transposes anywhere.
// Trick #2: compute S^T = K·Q^T so softmaxed P lands in C/D layout ==
//   B-fragment layout of out^T = vT·P: zero cross-lane movement for P.
// Trick #3: flash kernel is a 4-wave WG; K/V key-chunks are staged to LDS by
//   the Tensor Data Mover (tensor_load_to_lds, double-buffered, TENSORcnt),
//   giving 4x reuse and async global->LDS overlap with WMMA compute.
// ---------------------------------------------------------------------------

typedef __attribute__((ext_vector_type(16))) __bf16 v16bf;
typedef __attribute__((ext_vector_type(8)))  __bf16 v8bf;
typedef __attribute__((ext_vector_type(8)))  float  v8f;
typedef __attribute__((ext_vector_type(4)))  unsigned int u32x4;
typedef __attribute__((ext_vector_type(8)))  int i32x8;
typedef __attribute__((ext_vector_type(4)))  int i32x4;

#define C_DIM 1024
#define HEADS 16
#define HD    64
#define BATCH 4
#define NQ    512
#define MK    2048
#define SCL   0.125f      // 1/sqrt(64)

// Fragment loader for 16-bit A (row-major [16 x K tile]) and, identically,
// for B when the B source is column-major-as-row-major ([N rows x K]).
// lane<16: row=lane,     elems {k0..k0+7, k0+16..k0+23}
// lane>=16: row=lane-16, elems {k0+8..k0+15, k0+24..k0+31}
__device__ __forceinline__ v16bf load_frag(const __bf16* __restrict__ base,
                                           int stride, int lane) {
  const __bf16* p = base + (size_t)(lane & 15) * stride + ((lane >> 4) << 3);
  v16bf f;
  v8bf* h = reinterpret_cast<v8bf*>(&f);
  h[0] = *reinterpret_cast<const v8bf*>(p);
  h[1] = *reinterpret_cast<const v8bf*>(p + 16);
  return f;
}

__device__ __forceinline__ v8f wmma_bf16(v16bf a, v16bf b, v8f c) {
  return __builtin_amdgcn_wmma_f32_16x16x32_bf16(
      /*neg_a=*/false, a, /*neg_b=*/false, b,
      /*c_mod=*/(short)0, c, /*reuse_a=*/false, /*reuse_b=*/false);
}

// TDM: DMA a 2-D bf16 tile [rows x dim0] (row stride `stride` elements) from
// global into LDS at byte offset lds_off. Descriptor packing per ISA 08 §8.3/8.4.
// This toolchain exposes the 6-arg builtin: (g0, g1, g2, g3, g4, cpol).
__device__ __forceinline__ void tdm_load_2d(unsigned lds_off, const void* gptr,
                                            unsigned dim0, unsigned rows,
                                            unsigned stride) {
  const unsigned long long ga = (unsigned long long)gptr;
  u32x4 g0;
  g0[0] = 1u;                                               // count=1, flags=0
  g0[1] = lds_off;                                          // lds_addr
  g0[2] = (unsigned)ga;                                     // global_addr[31:0]
  g0[3] = (unsigned)((ga >> 32) & 0x1FFFFFFu) | 0x80000000u; // addr[56:32]|type=2
  i32x8 g1;
  g1[0] = 0x00010000;                                // data_size = 1 (2 bytes)
  g1[1] = (int)((dim0 & 0xFFFFu) << 16);             // tensor_dim0[15:0]
  g1[2] = (int)((dim0 >> 16) | ((rows & 0xFFFFu) << 16));   // dim0 hi | dim1 lo
  g1[3] = (int)((rows >> 16) | ((dim0 & 0xFFFFu) << 16));   // dim1 hi | tile_dim0
  g1[4] = (int)(rows & 0xFFFFu);                     // tile_dim1 (tile_dim2 = 0)
  g1[5] = (int)stride;                               // tensor_dim0_stride[31:0]
  g1[6] = 0;                                         // stride hi | dim1_stride lo
  g1[7] = 0;
  i32x4 z4; z4[0] = 0; z4[1] = 0; z4[2] = 0; z4[3] = 0;
  i32x8 z8;
#pragma unroll
  for (int i = 0; i < 8; ++i) z8[i] = 0;
  __builtin_amdgcn_tensor_load_to_lds(g0, g1, z4, z4, z8, 0);
}

// ------------------------------ prep kernels -------------------------------
__global__ void ca_f32_to_bf16(const float* __restrict__ in,
                               __bf16* __restrict__ out, int n) {
  int i = blockIdx.x * blockDim.x + threadIdx.x;
  if (i < n) out[i] = (__bf16)in[i];
}

__global__ void ca_mask_to_bias(const int* __restrict__ m,
                                float* __restrict__ out, int n) {
  int i = blockIdx.x * blockDim.x + threadIdx.x;
  if (i < n) out[i] = (m[i] != 0) ? 0.0f : -1e30f;
}

// ------------------------------ GEMM kernel --------------------------------
// Y[r,c] = sum_k A[r,k] * W[c,k] + bias[c].  One wave per 64x64 tile.
// MODE 0: bf16 row-major out.  MODE 1: bf16 head-transposed out vT[b,h,d,m].
// MODE 2: f32 row-major out (final projection).
template <int MODE>
__global__ __launch_bounds__(32) void ca_gemm(const __bf16* __restrict__ A,
                                              const __bf16* __restrict__ W,
                                              const float* __restrict__ bias,
                                              void* __restrict__ out, int K) {
  const int lane = threadIdx.x;
  const int r0 = blockIdx.x * 64;
  const int c0 = blockIdx.y * 64;

  v8f acc[4][4];
#pragma unroll
  for (int i = 0; i < 4; ++i)
#pragma unroll
    for (int j = 0; j < 4; ++j)
#pragma unroll
      for (int e = 0; e < 8; ++e) acc[i][j][e] = 0.0f;

  for (int kk = 0; kk < K; kk += 32) {
    v16bf a[4], w[4];
#pragma unroll
    for (int i = 0; i < 4; ++i)
      a[i] = load_frag(A + (size_t)(r0 + 16 * i) * K + kk, K, lane);
#pragma unroll
    for (int j = 0; j < 4; ++j)
      w[j] = load_frag(W + (size_t)(c0 + 16 * j) * K + kk, K, lane);
    if (kk + 32 < K) {  // gfx1250 global_prefetch_b8 for the next K slab
      __builtin_prefetch((const void*)(A + (size_t)(r0 + (lane & 15)) * K + kk + 32), 0, 3);
      __builtin_prefetch((const void*)(W + (size_t)(c0 + (lane & 15)) * K + kk + 32), 0, 3);
    }
#pragma unroll
    for (int i = 0; i < 4; ++i)
#pragma unroll
      for (int j = 0; j < 4; ++j)
        acc[i][j] = wmma_bf16(a[i], w[j], acc[i][j]);
  }

  // C/D layout: lane<16 -> col=lane, rows 0..7 ; lane>=16 -> col=lane-16, rows 8..15
  const int n  = lane & 15;
  const int mo = (lane >> 4) * 8;
#pragma unroll
  for (int i = 0; i < 4; ++i)
#pragma unroll
    for (int j = 0; j < 4; ++j) {
      const int c = c0 + 16 * j + n;
      const float bv = bias[c];
#pragma unroll
      for (int e = 0; e < 8; ++e) {
        const int r = r0 + 16 * i + mo + e;
        const float val = acc[i][j][e] + bv;
        if (MODE == 0) {
          ((__bf16*)out)[(size_t)r * C_DIM + c] = (__bf16)val;
        } else if (MODE == 2) {
          ((float*)out)[(size_t)r * C_DIM + c] = val;
        } else {  // vT[b, h, d, m]
          const int b = r >> 11, m = r & (MK - 1);
          const int h = c >> 6,  d = c & 63;
          ((__bf16*)out)[(((size_t)b * HEADS + h) * HD + d) * MK + m] = (__bf16)val;
        }
      }
    }
}

// --------------------------- flash attention -------------------------------
// 4 waves per WG; wave w owns 16 query rows (n0 = wg_n0 + 16w). Wave 0 drives
// the TDM: each 32-key chunk of K (32x64 bf16) and vT (64x32 bf16) is DMA'd
// into double-buffered LDS; all 4 waves consume it (4x reuse) while the next
// chunk streams in. S^T tiles keep softmax stats per-lane, and the exp'd P
// registers ARE the B fragment of out^T = vT · P.
__global__ __launch_bounds__(128) void ca_flash(const __bf16* __restrict__ qb,
                                                const __bf16* __restrict__ kb,
                                                const __bf16* __restrict__ vT,
                                                const float* __restrict__ mbias,
                                                __bf16* __restrict__ ob) {
  __shared__ __bf16 ldsK[2][32][64];   // [buf][m][d]   4KB each
  __shared__ __bf16 ldsV[2][64][32];   // [buf][d][m]   4KB each

  const int lane = threadIdx.x & 31;
  const int wid  = threadIdx.x >> 5;
  const int nt   = blockIdx.x & 7;            // N/64 = 8 tiles of 64 rows
  const int h    = (blockIdx.x >> 3) & 15;
  const int b    = blockIdx.x >> 7;
  const int n0   = nt * 64 + wid * 16;
  const int mo   = (lane >> 4) * 8;

  const __bf16* qbase = qb + (size_t)(b * NQ + n0) * C_DIM + h * HD;
  const v16bf qf0 = load_frag(qbase,      C_DIM, lane);   // d = 0..31
  const v16bf qf1 = load_frag(qbase + 32, C_DIM, lane);   // d = 32..63

  const __bf16* kbase = kb + (size_t)b * MK * C_DIM + h * HD;
  const __bf16* vbase = vT + (size_t)(b * HEADS + h) * HD * MK;
  const float*  mb    = mbias + b * MK;

  v8f o[4];
#pragma unroll
  for (int j = 0; j < 4; ++j)
#pragma unroll
    for (int e = 0; e < 8; ++e) o[j][e] = 0.0f;

  float run_m = -1e30f, run_s = 0.0f;

  // Prologue: DMA chunk 0 into buffer 0.
  if (wid == 0) {
    tdm_load_2d((unsigned)(size_t)&ldsK[0][0][0], kbase, HD, 32, C_DIM);
    tdm_load_2d((unsigned)(size_t)&ldsV[0][0][0], vbase, 32, HD, MK);
    __builtin_amdgcn_s_wait_tensorcnt(0);
  }
  __syncthreads();

  for (int m0 = 0; m0 < MK; m0 += 32) {
    const int buf = (m0 >> 5) & 1;
    if (wid == 0 && (m0 + 32) < MK) {  // prefetch next chunk into other buffer
      tdm_load_2d((unsigned)(size_t)&ldsK[buf ^ 1][0][0],
                  kbase + (size_t)(m0 + 32) * C_DIM, HD, 32, C_DIM);
      tdm_load_2d((unsigned)(size_t)&ldsV[buf ^ 1][0][0],
                  vbase + (m0 + 32), 32, HD, MK);
    }

    const __bf16* kp = &ldsK[buf][0][0];
    const v16bf k00 = load_frag(kp,             HD, lane);
    const v16bf k01 = load_frag(kp + 32,        HD, lane);
    const v16bf k10 = load_frag(kp + 16 * HD,      HD, lane);
    const v16bf k11 = load_frag(kp + 16 * HD + 32, HD, lane);

    v8f s0, s1;
#pragma unroll
    for (int e = 0; e < 8; ++e) { s0[e] = 0.0f; s1[e] = 0.0f; }
    s0 = wmma_bf16(k00, qf0, s0);  s0 = wmma_bf16(k01, qf1, s0);
    s1 = wmma_bf16(k10, qf0, s1);  s1 = wmma_bf16(k11, qf1, s1);

    float pm = -1e30f;
#pragma unroll
    for (int e = 0; e < 8; ++e) {
      s0[e] = s0[e] * SCL + mb[m0 + mo + e];
      s1[e] = s1[e] * SCL + mb[m0 + 16 + mo + e];
      pm = fmaxf(pm, fmaxf(s0[e], s1[e]));
    }
    pm = fmaxf(pm, __shfl_xor(pm, 16));
    const float new_m = fmaxf(run_m, pm);
    const float corr  = __expf(run_m - new_m);
    run_m = new_m;

    float ps = 0.0f;
    v16bf pf;  // B fragment of out^T = vT · P, built in-register
#pragma unroll
    for (int e = 0; e < 8; ++e) {
      const float p0 = __expf(s0[e] - new_m);
      const float p1 = __expf(s1[e] - new_m);
      ps += p0 + p1;
      pf[e]     = (__bf16)p0;   // K rows m0+{0..7|8..15}
      pf[e + 8] = (__bf16)p1;   // K rows m0+16+{0..7|8..15}
    }
    ps += __shfl_xor(ps, 16);
    run_s = run_s * corr + ps;

#pragma unroll
    for (int j = 0; j < 4; ++j)
#pragma unroll
      for (int e = 0; e < 8; ++e) o[j][e] *= corr;

#pragma unroll
    for (int jd = 0; jd < 4; ++jd) {
      const v16bf vf = load_frag(&ldsV[buf][jd * 16][0], 32, lane);
      o[jd] = wmma_bf16(vf, pf, o[jd]);
    }

    if (wid == 0) __builtin_amdgcn_s_wait_tensorcnt(0);
    __syncthreads();  // next buffer ready; this buffer free for next DMA
  }

  // out^T C/D layout: col n = lane&15, rows d = jd*16 + mo + e -> contiguous
  const float inv = 1.0f / run_s;
  __bf16* op = ob + (size_t)(b * NQ + n0 + (lane & 15)) * C_DIM + h * HD + mo;
#pragma unroll
  for (int jd = 0; jd < 4; ++jd) {
    v8bf ov;
#pragma unroll
    for (int e = 0; e < 8; ++e) ov[e] = (__bf16)(o[jd][e] * inv);
    *reinterpret_cast<v8bf*>(op + jd * 16) = ov;
  }
}

// ------------------------------- launcher ----------------------------------
extern "C" void kernel_launch(void* const* d_in, const int* in_sizes, int n_in,
                              void* d_out, int out_size, void* d_ws, size_t ws_size,
                              hipStream_t stream) {
  const float* x    = (const float*)d_in[0];
  const float* ctx  = (const float*)d_in[1];
  const int*   mask = (const int*)d_in[2];
  const float* Wq = (const float*)d_in[3];  const float* bq = (const float*)d_in[4];
  const float* Wk = (const float*)d_in[5];  const float* bk = (const float*)d_in[6];
  const float* Wv = (const float*)d_in[7];  const float* bv = (const float*)d_in[8];
  const float* Wo = (const float*)d_in[9];  const float* bo = (const float*)d_in[10];
  float* out = (float*)d_out;

  char* ws = (char*)d_ws;
  size_t off = 0;
  auto take = [&](size_t bytes) -> void* {
    void* p = ws + off;
    off += (bytes + 255) & ~(size_t)255;
    return p;
  };
  const size_t NX = (size_t)BATCH * NQ * C_DIM;   // 2M
  const size_t NC = (size_t)BATCH * MK * C_DIM;   // 8M
  const size_t NW = (size_t)C_DIM * C_DIM;        // 1M
  __bf16* xb  = (__bf16*)take(NX * 2);
  __bf16* cb  = (__bf16*)take(NC * 2);
  __bf16* wqb = (__bf16*)take(NW * 2);
  __bf16* wkb = (__bf16*)take(NW * 2);
  __bf16* wvb = (__bf16*)take(NW * 2);
  __bf16* wob = (__bf16*)take(NW * 2);
  __bf16* q_b = (__bf16*)take(NX * 2);
  __bf16* k_b = (__bf16*)take(NC * 2);
  __bf16* vTb = (__bf16*)take(NC * 2);
  __bf16* o_b = (__bf16*)take(NX * 2);
  float*  mbb = (float*)take((size_t)BATCH * MK * 4);

  auto cvt = [&](const float* in, __bf16* o, int n) {
    ca_f32_to_bf16<<<(n + 255) / 256, 256, 0, stream>>>(in, o, n);
  };
  cvt(x,   xb,  (int)NX);
  cvt(ctx, cb,  (int)NC);
  cvt(Wq,  wqb, (int)NW);
  cvt(Wk,  wkb, (int)NW);
  cvt(Wv,  wvb, (int)NW);
  cvt(Wo,  wob, (int)NW);
  ca_mask_to_bias<<<(BATCH * MK + 255) / 256, 256, 0, stream>>>(mask, mbb, BATCH * MK);

  const dim3 gq(BATCH * NQ / 64, C_DIM / 64);   // (32, 16)
  const dim3 gc(BATCH * MK / 64, C_DIM / 64);   // (128, 16)
  ca_gemm<0><<<gq, 32, 0, stream>>>(xb, wqb, bq, (void*)q_b, C_DIM);  // Q
  ca_gemm<0><<<gc, 32, 0, stream>>>(cb, wkb, bk, (void*)k_b, C_DIM);  // K
  ca_gemm<1><<<gc, 32, 0, stream>>>(cb, wvb, bv, (void*)vTb, C_DIM);  // V (head-T)

  ca_flash<<<BATCH * HEADS * (NQ / 64), 128, 0, stream>>>(q_b, k_b, vTb, mbb, o_b);

  ca_gemm<2><<<gq, 32, 0, stream>>>(o_b, wob, bo, (void*)out, C_DIM); // O proj
}
